// DSSMamba_6588479832210
// MI455X (gfx1250) — matrix operand order repaired and checked by reading, not compiled, not run on hardware
//
#include <hip/hip_runtime.h>
#include <hip/hip_bf16.h>

// ---------------------------------------------------------------------------
// DSSMamba (bidirectional Mamba block) for MI455X / gfx1250, wave32 + WMMA.
// GEMMs: v_wmma_f32_16x16x32_bf16 (fp32 accumulate), B-operand staged to LDS
// via the Tensor Data Mover (tensor_load_to_lds + s_wait_tensorcnt).
// Scan: register-resident 16-state recurrence, one lane per (b,d) channel.
// ---------------------------------------------------------------------------

typedef __bf16 bf16_t;
typedef __bf16 v16bf __attribute__((ext_vector_type(16)));
typedef float  v8f   __attribute__((ext_vector_type(8)));
typedef unsigned int u32x4 __attribute__((ext_vector_type(4)));
typedef int          i32x4 __attribute__((ext_vector_type(4)));
typedef int          i32x8 __attribute__((ext_vector_type(8)));

#define BSZ   4
#define LSEQ  4096
#define DMODEL 256
#define DINNER 512
#define BL    (BSZ * LSEQ)   // 16384

__device__ __forceinline__ unsigned short f2bf(float f) {
    unsigned int x = __float_as_uint(f);
    unsigned int r = (x + 0x7FFFu + ((x >> 16) & 1u)) >> 16;   // RNE
    return (unsigned short)r;
}
__device__ __forceinline__ unsigned int pack2(float a, float b) {
    return (unsigned int)f2bf(a) | ((unsigned int)f2bf(b) << 16);
}
__device__ __forceinline__ float silu_f(float x) {
    return x * (1.0f / (1.0f + __expf(-x)));
}

union Frag16 { uint4 q[2]; v16bf v; };

// ---------------------------------------------------------------------------
// TDM: DMA a 2D bf16 tile (tile_d1 rows x tile_d0 cols, row stride stride0
// elements) from global memory into LDS at lds_addr. Issue from ONE wave.
// Descriptor layout per CDNA5 ISA ch.8 (D# groups 0/1; groups 2/3 unused->0).
// ---------------------------------------------------------------------------
__device__ __forceinline__ void tdm_load_tile_bf16(
    const void* gptr, unsigned lds_addr,
    unsigned tile_d0, unsigned tile_d1, unsigned long long stride0) {
    unsigned long long ga = (unsigned long long)(uintptr_t)gptr;
    u32x4 g0;
    g0[0] = 1u;                                         // count=1 (user D#)
    g0[1] = lds_addr;                                   // LDS byte address
    g0[2] = (unsigned)(ga & 0xFFFFFFFFu);               // global_addr[31:0]
    g0[3] = (unsigned)((ga >> 32) & 0x1FFFFFFu)         // global_addr[56:32]
          | (2u << 30);                                 // type=2 ("image")
    i32x8 g1;
    g1[0] = (int)(1u << 16);                            // data_size=1 -> 2B
    g1[1] = (int)((tile_d0 & 0xFFFFu) << 16);           // tensor_dim0 = tile_d0
    g1[2] = (int)((tile_d0 >> 16) | ((tile_d1 & 0xFFFFu) << 16)); // tensor_dim1
    g1[3] = (int)((tile_d1 >> 16) | (tile_d0 << 16));   // tile_dim0
    g1[4] = (int)(tile_d1);                             // tile_dim1 (tile_dim2=0)
    g1[5] = (int)(unsigned)(stride0 & 0xFFFFFFFFu);     // dim0 stride lo
    g1[6] = (int)(unsigned)((stride0 >> 32) & 0xFFFFu); // dim0 stride hi
    g1[7] = 0;                                          // dim1 stride (unused)
    i32x4 gz = {0, 0, 0, 0};
#if __clang_major__ >= 23
    i32x8 gz8 = {0, 0, 0, 0, 0, 0, 0, 0};
    __builtin_amdgcn_tensor_load_to_lds(g0, g1, gz, gz, gz8, 0);
#else
    __builtin_amdgcn_tensor_load_to_lds(g0, g1, gz, gz, 0);
#endif
}

// ---------------------------------------------------------------------------
// Weight prep: W (Nr, Kc) fp32 row-major  ->  out (Kc, Nr) bf16 row-major
// ---------------------------------------------------------------------------
__global__ void wprep_kernel(const float* __restrict__ W,
                             unsigned short* __restrict__ out,
                             int Nr, int Kc) {
    int idx = blockIdx.x * 256 + threadIdx.x;
    if (idx >= Nr * Kc) return;
    int k = idx / Nr;
    int n = idx - k * Nr;
    out[idx] = f2bf(W[(size_t)n * Kc + k]);
}

// ---------------------------------------------------------------------------
// Generic bf16 WMMA GEMM:  C(M,N) = A(M,K) * Bw(K,N)   (+ optional SiLU on A)
//   A   : fp32, row stride lda, converted to bf16 while staging to LDS
//   Bw  : bf16 (pre-converted, K x N row-major, row stride ldb), staged by TDM
//   Block tile BM x BN, K-step 32, WM x WN waves, each wave TM x TN 16x16 tiles
// Fragment layouts per CDNA5 ISA 7.12.2 (wave32).
// ---------------------------------------------------------------------------
template<int BM, int BN, int WM, int WN, int TM, int TN, bool SILU>
__global__ __launch_bounds__(WM * WN * 32)
void gemm_bf16_wmma(const float* __restrict__ A, int lda,
                    const unsigned short* __restrict__ Bw, int ldb,
                    float* __restrict__ C, int ldc,
                    int M, int N, int K) {
    constexpr int KS = 32;
    constexpr int NT = WM * WN * 32;
    __shared__ __align__(16) unsigned short As[BM * KS];  // [m][k], 64B rows
    __shared__ __align__(16) unsigned short Bs[KS * BN];  // [k][n]

    const int tid  = threadIdx.x;
    const int lane = tid & 31;
    const int w    = tid >> 5;
    const int wm   = w / WN;
    const int wn   = w - wm * WN;
    const unsigned n0 = blockIdx.x * BN;
    const unsigned m0 = blockIdx.y * BM;
    const int mrow = lane & 15;
    const int half = lane >> 4;
    const unsigned bs_lds = (unsigned)(uintptr_t)(void*)Bs;  // LDS byte offset

    v8f acc[TM][TN] = {};

    for (int kk = 0; kk < K; kk += KS) {
        // ---- stage B via Tensor Data Mover (wave 0 only; EXEC ignored) ----
        if (w == 0) {
            tdm_load_tile_bf16(Bw + (size_t)kk * ldb + n0, bs_lds,
                               (unsigned)BN, (unsigned)KS,
                               (unsigned long long)ldb);
        }
        // ---- stage A (fp32 -> bf16, optional SiLU) ------------------------
        for (int i = tid; i < BM * (KS / 4); i += NT) {
            int r  = i >> 3;           // KS/4 == 8
            int c4 = i & 7;
            unsigned aoff = (m0 + (unsigned)r) * (unsigned)lda
                          + (unsigned)(kk + c4 * 4);
            const float4 av = *(const float4*)(A + aoff);
            float x0 = av.x, x1 = av.y, x2 = av.z, x3 = av.w;
            if (SILU) { x0 = silu_f(x0); x1 = silu_f(x1);
                        x2 = silu_f(x2); x3 = silu_f(x3); }
            unsigned int* dst = (unsigned int*)As + r * (KS / 2) + c4 * 2;
            dst[0] = pack2(x0, x1);
            dst[1] = pack2(x2, x3);
            if (kk + KS < K)  // hint next K-slab into cache (global_prefetch)
                __builtin_prefetch(A + aoff + KS, 0, 3);
        }
        __builtin_amdgcn_s_wait_tensorcnt(0);   // no-op for waves 1..N
        __syncthreads();

        // ---- fragments + WMMA ---------------------------------------------
        Frag16 af[TM], bf[TN];
#pragma unroll
        for (int i = 0; i < TM; i++) {
            int mr = (wm * TM + i) * 16 + mrow;   // row in As (64B stride)
            const char* base = (const char*)As + mr * 64 + half * 16;
            af[i].q[0] = *(const uint4*)(base);
            af[i].q[1] = *(const uint4*)(base + 32);
        }
#pragma unroll
        for (int j = 0; j < TN; j++) {
            int nb = (wn * TN + j) * 16;          // lane%16 == n col of Bs, elements over k
            const unsigned short* bcol = Bs + nb + mrow;
            unsigned int bw[8];
#pragma unroll
            for (int p = 0; p < 8; p++) {
                int k = (p < 4) ? (8 * half + 2 * p) : (16 + 8 * half + 2 * (p - 4));
                bw[p] = (unsigned int)bcol[k * BN] | ((unsigned int)bcol[(k + 1) * BN] << 16);
            }
            bf[j].q[0] = make_uint4(bw[0], bw[1], bw[2], bw[3]);
            bf[j].q[1] = make_uint4(bw[4], bw[5], bw[6], bw[7]);
        }
#pragma unroll
        for (int i = 0; i < TM; i++)
#pragma unroll
            for (int j = 0; j < TN; j++)
                acc[i][j] = __builtin_amdgcn_wmma_f32_16x16x32_bf16(
                    false, af[i].v, false, bf[j].v,
                    (short)0, acc[i][j], false, false);
        __syncthreads();
    }

    // ---- epilogue: C layout lanes 0-15: M=v,N=lane; 16-31: M=8+v ----------
#pragma unroll
    for (int i = 0; i < TM; i++) {
        unsigned mb = m0 + (unsigned)((wm * TM + i) * 16 + half * 8);
#pragma unroll
        for (int j = 0; j < TN; j++) {
            unsigned cn = n0 + (unsigned)((wn * TN + j) * 16 + mrow);
            unsigned cbase = mb * (unsigned)ldc + cn;
#pragma unroll
            for (int v = 0; v < 8; v++)
                C[cbase + (unsigned)(v * ldc)] = acc[i][j][v];
        }
    }
}

// ---------------------------------------------------------------------------
// Column-mean partial sums over L (xz is (B, L, 2048)); mean pre-zeroed.
// ---------------------------------------------------------------------------
__global__ void mean_partial_kernel(const float* __restrict__ xz,
                                    float* __restrict__ mean) {
    int f  = blockIdx.x * 256 + threadIdx.x;   // 0..2047
    int b  = blockIdx.z;
    int l0 = blockIdx.y * 512;
    const float* p = xz + ((size_t)b * LSEQ + l0) * 2048 + f;
    float s = 0.f;
    for (int i = 0; i < 512; i++) s += p[(size_t)i * 2048];
    atomicAdd(&mean[b * 2048 + f], s);
}

// gate[b,f] = b_g[f] + (mean[b,:]/L) . W_g[f,:]
__global__ void gate_kernel(const float* __restrict__ mean,
                            const float* __restrict__ W_g,
                            const float* __restrict__ b_g,
                            float* __restrict__ gate) {
    __shared__ float mrow[2048];
    int b = blockIdx.y;
    int f = blockIdx.x * 256 + threadIdx.x;
    for (int i = threadIdx.x; i < 2048; i += 256)
        mrow[i] = mean[b * 2048 + i] * (1.0f / LSEQ);
    __syncthreads();
    const float* wr = W_g + (size_t)f * 2048;
    float s = b_g[f];
    for (int j = 0; j < 2048; j += 4)
        s += mrow[j] * wr[j] + mrow[j + 1] * wr[j + 1] +
             mrow[j + 2] * wr[j + 2] + mrow[j + 3] * wr[j + 3];
    gate[b * 2048 + f] = s;
}

__global__ void addgate_kernel(float* __restrict__ xz,
                               const float* __restrict__ gate) {
    size_t idx = (size_t)blockIdx.x * 256 + threadIdx.x;  // float4 units
    size_t e   = idx * 4;
    size_t row = e >> 11;            // / 2048
    int    f   = (int)(e & 2047);
    int    b   = (int)(row >> 12);   // / LSEQ
    float4 g = *(const float4*)(gate + b * 2048 + f);
    float4* p = (float4*)(xz + row * 2048 + f);
    float4 v = *p;
    v.x += g.x; v.y += g.y; v.z += g.z; v.w += g.w;
    *p = v;
}

// ---------------------------------------------------------------------------
// dt = softplus(dt_low @ W_dt.T + b_dt); dt_low = xdbl[:, :, 0:16]
// one block per (b,l) row, one thread per d.
// ---------------------------------------------------------------------------
__global__ void dt_kernel(const float* __restrict__ xdbl,
                          const float* __restrict__ W_dt,
                          const float* __restrict__ b_dt,
                          float* __restrict__ dtout) {
    __shared__ float lo[16];
    int row = blockIdx.x;
    int d   = threadIdx.x;   // 0..511
    if (d < 16) lo[d] = xdbl[(size_t)row * 48 + d];
    __syncthreads();
    const float4* wv = (const float4*)(W_dt + (size_t)d * 16);
    float4 w0 = wv[0], w1 = wv[1], w2 = wv[2], w3 = wv[3];
    float acc = b_dt[d];
    acc += lo[0]  * w0.x + lo[1]  * w0.y + lo[2]  * w0.z + lo[3]  * w0.w;
    acc += lo[4]  * w1.x + lo[5]  * w1.y + lo[6]  * w1.z + lo[7]  * w1.w;
    acc += lo[8]  * w2.x + lo[9]  * w2.y + lo[10] * w2.z + lo[11] * w2.w;
    acc += lo[12] * w3.x + lo[13] * w3.y + lo[14] * w3.z + lo[15] * w3.w;
    float sp = (acc > 20.f) ? acc : log1pf(__expf(acc));
    dtout[(size_t)row * 512 + d] = sp;
}

// ---------------------------------------------------------------------------
// Selective scan: one lane per (b,d) channel, 16 states in registers.
// u = silu(x) recomputed on the fly; y = sum_n h[n]*C[n] + u*D.
// ---------------------------------------------------------------------------
__global__ void scan_kernel(const float* __restrict__ xz, int xcol,
                            const float* __restrict__ dtbuf,
                            const float* __restrict__ xdbl,
                            const float* __restrict__ A_log,
                            const float* __restrict__ Dvec,
                            float* __restrict__ y,
                            int reverse) {
    int c = blockIdx.x * 256 + threadIdx.x;   // 0..2047
    int b = c >> 9;
    int d = c & 511;

    float An[16];
#pragma unroll
    for (int n = 0; n < 16; n++) An[n] = -__expf(A_log[d * 16 + n]);
    float Dd = Dvec[d];
    float h[16];
#pragma unroll
    for (int n = 0; n < 16; n++) h[n] = 0.f;

    const float* xb  = xz    + (size_t)b * LSEQ * 2048 + xcol + d;
    const float* dtb = dtbuf + (size_t)b * LSEQ * 512 + d;
    const float* bcb = xdbl  + (size_t)b * LSEQ * 48;
    float*       yb  = y     + (size_t)b * LSEQ * 512 + d;

    for (int s = 0; s < LSEQ; s++) {
        int t = reverse ? (LSEQ - 1 - s) : s;
        float dt = dtb[(size_t)t * 512];
        float xv = xb[(size_t)t * 2048];
        float u  = silu_f(xv);
        float du = dt * u;
        const float* bc = bcb + (size_t)t * 48;
        float yv = 0.f;
#pragma unroll
        for (int n = 0; n < 16; n++) {
            float dA = __expf(dt * An[n]);
            h[n] = fmaf(h[n], dA, du * bc[16 + n]);
            yv   = fmaf(h[n], bc[32 + n], yv);
        }
        yb[(size_t)t * 512] = fmaf(u, Dd, yv);
    }
}

// yg[row, 0:512] = y_f * silu(z_f);  yg[row, 512:1024] = y_b * silu(z_b)
__global__ void ygate_kernel(const float* __restrict__ y_f,
                             const float* __restrict__ y_b,
                             const float* __restrict__ xz,
                             float* __restrict__ yg) {
    size_t i   = (size_t)blockIdx.x * 256 + threadIdx.x;  // BL*512
    size_t row = i >> 9;
    int    d   = (int)(i & 511);
    float zf = xz[row * 2048 + 512  + d];
    float zb = xz[row * 2048 + 1536 + d];
    yg[row * 1024 + d]       = y_f[row * 512 + d] * silu_f(zf);
    yg[row * 1024 + 512 + d] = y_b[row * 512 + d] * silu_f(zb);
}

// ---------------------------------------------------------------------------
extern "C" void kernel_launch(void* const* d_in, const int* in_sizes, int n_in,
                              void* d_out, int out_size, void* d_ws, size_t ws_size,
                              hipStream_t stream) {
    (void)in_sizes; (void)n_in; (void)out_size; (void)ws_size;

    const float* hidden  = (const float*)d_in[0];
    const float* W_in    = (const float*)d_in[1];
    const float* W_g     = (const float*)d_in[2];
    const float* b_g     = (const float*)d_in[3];
    const float* W_xp_f  = (const float*)d_in[4];
    const float* W_xp_b  = (const float*)d_in[5];
    const float* W_dt_f  = (const float*)d_in[6];
    const float* b_dt_f  = (const float*)d_in[7];
    const float* W_dt_b  = (const float*)d_in[8];
    const float* b_dt_b  = (const float*)d_in[9];
    const float* A_log_f = (const float*)d_in[10];
    const float* A_log_b = (const float*)d_in[11];
    const float* D_f     = (const float*)d_in[12];
    const float* D_b     = (const float*)d_in[13];
    const float* W_out   = (const float*)d_in[14];

    char* ws = (char*)d_ws;
    size_t off = 0;
    auto alloc = [&](size_t bytes) {
        void* p = ws + off;
        off += (bytes + 255) & ~(size_t)255;
        return p;
    };
    float* xz      = (float*)alloc((size_t)BL * 2048 * 4);  // 134 MB
    float* yg      = (float*)alloc((size_t)BL * 1024 * 4);  //  67 MB
    float* dt_f    = (float*)alloc((size_t)BL * 512 * 4);   //  33 MB
    float* dt_b    = (float*)alloc((size_t)BL * 512 * 4);
    float* y_f     = (float*)alloc((size_t)BL * 512 * 4);
    float* y_b     = (float*)alloc((size_t)BL * 512 * 4);
    float* xdbl_f  = (float*)alloc((size_t)BL * 48 * 4);
    float* xdbl_b  = (float*)alloc((size_t)BL * 48 * 4);
    unsigned short* BwIn  = (unsigned short*)alloc((size_t)256  * 2048 * 2);
    unsigned short* BwXpF = (unsigned short*)alloc((size_t)512  * 48   * 2);
    unsigned short* BwXpB = (unsigned short*)alloc((size_t)512  * 48   * 2);
    unsigned short* BwOut = (unsigned short*)alloc((size_t)1024 * 256  * 2);
    float* meanb   = (float*)alloc((size_t)BSZ * 2048 * 4);
    float* gateb   = (float*)alloc((size_t)BSZ * 2048 * 4);

    // 0) weight prep (transpose + bf16 convert)
    wprep_kernel<<<(2048 * 256 + 255) / 256, 256, 0, stream>>>(W_in,  BwIn, 2048, 256);
    wprep_kernel<<<(48 * 512 + 255) / 256,   256, 0, stream>>>(W_xp_f, BwXpF, 48, 512);
    wprep_kernel<<<(48 * 512 + 255) / 256,   256, 0, stream>>>(W_xp_b, BwXpB, 48, 512);
    wprep_kernel<<<(256 * 1024 + 255) / 256, 256, 0, stream>>>(W_out, BwOut, 256, 1024);

    // 1) xz = hidden @ W_in.T  (M=16384, N=2048, K=256)
    gemm_bf16_wmma<128, 128, 2, 4, 4, 2, false>
        <<<dim3(2048 / 128, BL / 128), 256, 0, stream>>>(
            hidden, 256, BwIn, 2048, xz, 2048, BL, 2048, 256);

    // 2) mean over L, gate, broadcast add
    hipMemsetAsync(meanb, 0, (size_t)BSZ * 2048 * 4, stream);
    mean_partial_kernel<<<dim3(8, LSEQ / 512, BSZ), 256, 0, stream>>>(xz, meanb);
    gate_kernel<<<dim3(8, BSZ), 256, 0, stream>>>(meanb, W_g, b_g, gateb);
    addgate_kernel<<<(BL * 2048 / 4) / 256, 256, 0, stream>>>(xz, gateb);

    // 3) xdbl = silu(x) @ W_xp.T  (M=16384, N=48, K=512) per branch
    gemm_bf16_wmma<128, 48, 8, 1, 1, 3, true>
        <<<dim3(1, BL / 128), 256, 0, stream>>>(
            xz + 0,    2048, BwXpF, 48, xdbl_f, 48, BL, 48, 512);
    gemm_bf16_wmma<128, 48, 8, 1, 1, 3, true>
        <<<dim3(1, BL / 128), 256, 0, stream>>>(
            xz + 1024, 2048, BwXpB, 48, xdbl_b, 48, BL, 48, 512);

    // 4) dt = softplus(dt_low @ W_dt.T + b_dt)
    dt_kernel<<<BL, 512, 0, stream>>>(xdbl_f, W_dt_f, b_dt_f, dt_f);
    dt_kernel<<<BL, 512, 0, stream>>>(xdbl_b, W_dt_b, b_dt_b, dt_b);

    // 5) selective scans (fwd on x_f, bwd on x_b)
    scan_kernel<<<8, 256, 0, stream>>>(xz, 0,    dt_f, xdbl_f, A_log_f, D_f, y_f, 0);
    scan_kernel<<<8, 256, 0, stream>>>(xz, 1024, dt_b, xdbl_b, A_log_b, D_b, y_b, 1);

    // 6) yg = [y_f*silu(z_f), y_b*silu(z_b)]
    ygate_kernel<<<(BL * 512) / 256, 256, 0, stream>>>(y_f, y_b, xz, yg);

    // 7) out = yg @ W_out.T  (M=16384, N=256, K=1024)
    gemm_bf16_wmma<128, 128, 2, 4, 4, 2, false>
        <<<dim3(2, BL / 128), 256, 0, stream>>>(
            yg, 1024, BwOut, 256, (float*)d_out, 256, BL, 256, 1024);
}
